// DyGMamba_90735479095629
// MI455X (gfx1250) — matrix operand order repaired and checked by reading, not compiled, not run on hardware
//
#include <hip/hip_runtime.h>
#include <hip/hip_bf16.h>
#include <math.h>

// ---------------------------------------------------------------------------
// DyGMamba forward for MI455X (gfx1250, wave32, WMMA).
// All dense GEMMs use v_wmma_f32_16x16x32_f16 (f16 A/B, f32 accumulate).
// Weights are pre-cast/transposed to f16 (Npad x Kpad) so both A and B
// fragments are contiguous 16B loads matching the ISA VGPR layouts.
// Input order assumed: setup_inputs() dict insertion order, params flattened
// depth-first in insertion order (mamba list layer0 then layer1).
// ---------------------------------------------------------------------------

typedef __attribute__((ext_vector_type(16))) _Float16 v16h;
typedef __attribute__((ext_vector_type(8)))  _Float16 v8h;
typedef __attribute__((ext_vector_type(8)))  float    v8f;

#define DEVI __device__ __forceinline__

DEVI float gelu_tanh(float x) {
  float x3 = x * x * x;
  return 0.5f * x * (1.0f + tanhf(0.7978845608028654f * (x + 0.044715f * x3)));
}
DEVI float softplus_f(float x) { return (x > 20.0f) ? x : log1pf(expf(x)); }
DEVI float silu_f(float x)     { return x / (1.0f + expf(-x)); }

// ------------------------------ WMMA GEMM ----------------------------------
// C[M x N] = act(A[M x K] @ W + bias).  A: row-major f16, lda == K (padded
// to mult of 32, zero-filled).  WT: f16 W transposed, (Npad x K) row-major,
// Npad = 16*Ntiles, zero-filled.  One wave per 16x16 output tile.
// ACT: 0=none 1=gelu(tanh) 2=softplus.  OMODE: 0=f32 store, 1=f32 +=, 2=f16.
template<int ACT, int OMODE>
__global__ __launch_bounds__(256) void gemm_wmma(
    const _Float16* __restrict__ A, const _Float16* __restrict__ WT, int K,
    const float* __restrict__ bias, float* __restrict__ Cf,
    _Float16* __restrict__ Ch, int ldc, int col0, int M, int N,
    int Mtiles, int Ntiles)
{
  const int lane = threadIdx.x & 31;
  const int wid  = threadIdx.x >> 5;
  int tile = blockIdx.x * (blockDim.x >> 5) + wid;
  if (tile >= Mtiles * Ntiles) return;          // wave-uniform: EXEC stays full
  const int tm = tile / Ntiles, tn = tile % Ntiles;
  const int nl   = lane & 15;                   // A row within tile / B,C col
  const int half = lane >> 4;
  // A frag: lane holds row m, K halves [koff..koff+7],[koff+16..koff+23]
  const _Float16* ap = A  + (size_t)(tm * 16 + nl) * K + half * 8;
  // B frag: lane holds col n, K halves [koff..koff+7],[koff+16..koff+23] in WT row n
  const _Float16* bp = WT + (size_t)(tn * 16 + nl) * K + half * 8;
  v8f acc = {};
  const int ksteps = K >> 5;
  for (int k = 0; k < ksteps; ++k) {
    union { v16h v; v8h h[2]; } ua, ub;
    ua.h[0] = *(const v8h*)(ap);
    ua.h[1] = *(const v8h*)(ap + 16);
    ub.h[0] = *(const v8h*)(bp);
    ub.h[1] = *(const v8h*)(bp + 16);
    __builtin_prefetch((const void*)(ap + 128), 0, 1);   // global_prefetch_b8
    acc = __builtin_amdgcn_wmma_f32_16x16x32_f16(false, ua.v, false, ub.v,
                                                 (short)0, acc, false, false);
    ap += 32; bp += 32;
  }
  const int n_out = tn * 16 + nl;
  if (n_out >= N) return;
  const float bv = bias ? bias[n_out] : 0.0f;
  const int mbase = tm * 16 + half * 8;         // C: VGPR r -> row r + 8*half
#pragma unroll
  for (int r = 0; r < 8; ++r) {
    const int m_out = mbase + r;
    if (m_out >= M) break;
    float v = acc[r] + bv;
    if (ACT == 1) v = gelu_tanh(v);
    if (ACT == 2) v = softplus_f(v);
    const size_t o = (size_t)m_out * ldc + col0 + n_out;
    if (OMODE == 0) Cf[o] = v;
    if (OMODE == 1) Cf[o] += v;
    if (OMODE == 2) Ch[o] = (_Float16)v;
  }
}

// --------------------- weight cast+transpose to f16 ------------------------
__global__ void k_prep_wt(const float* __restrict__ W, _Float16* __restrict__ WT,
                          int K, int N, int Kp, int Np) {
  int i = blockIdx.x * blockDim.x + threadIdx.x;
  if (i >= Np * Kp) return;
  int n = i / Kp, k = i % Kp;
  float v = (n < N && k < K) ? W[(size_t)k * N + n] : 0.0f;
  WT[i] = (_Float16)v;
}

// ------------------------- gather + cast (node/edge) -----------------------
__global__ void k_gather_cast(const float* __restrict__ feats,
                              const int* __restrict__ ids,
                              _Float16* __restrict__ out, int D, int ld, int nrow) {
  int i = blockIdx.x * blockDim.x + threadIdx.x;
  if (i >= nrow * ld) return;
  int row = i / ld, c = i % ld;
  float v = (c < D) ? feats[(size_t)ids[row] * D + c] : 0.0f;
  out[i] = (_Float16)v;
}

// ------------------------------ time features ------------------------------
__global__ void k_time(const float* __restrict__ t_now,
                       const float* __restrict__ nb_t,
                       const float* __restrict__ tw,
                       _Float16* __restrict__ out, int nrow) {
  int i = blockIdx.x * blockDim.x + threadIdx.x;
  if (i >= nrow * 128) return;
  int row = i / 128, c = i % 128;
  float v = 0.0f;
  if (c < 100) v = cosf((t_now[row / 512] - nb_t[row]) * tw[c]);
  out[i] = (_Float16)v;
}

// ------------------------- NIF co-occurrence counts ------------------------
__global__ __launch_bounds__(512) void k_nif_counts(
    const int* __restrict__ src_nb, const int* __restrict__ dst_nb,
    const int* __restrict__ src_ids, const int* __restrict__ dst_ids,
    float* __restrict__ app_src, float* __restrict__ app_dst) {
  __shared__ int ss[512], sd[512];
  const int b = blockIdx.x, i = threadIdx.x;
  const int s = src_nb[b * 512 + i];
  const int d = dst_nb[b * 512 + i];
  ss[i] = s; sd[i] = d;
  __syncthreads();
  int css = 0, csd = 0, cdd = 0, cds = 0;
  for (int j = 0; j < 512; ++j) {
    int sj = ss[j], dj = sd[j];
    css += (s == sj); csd += (s == dj);
    cdd += (d == dj); cds += (d == sj);
  }
  size_t row = (size_t)b * 512 + i;
  app_src[row * 2 + 0] = (float)css;
  app_src[row * 2 + 1] = (s == dst_ids[b]) ? (float)css : (float)csd;
  app_dst[row * 2 + 0] = (d == src_ids[b]) ? (float)cdd : (float)cds;
  app_dst[row * 2 + 1] = (float)cdd;
}

// --------------------- NIF encoder MLP (50-wide, 2 taps) -------------------
__global__ __launch_bounds__(256) void k_nif_enc(
    const float* __restrict__ app, const float* __restrict__ w1,
    const float* __restrict__ b1, const float* __restrict__ w2,
    const float* __restrict__ b2, _Float16* __restrict__ out, int nrow) {
  __shared__ float s_w2[2500];
  __shared__ float s_w1[50], s_b1[50], s_b2[50];
  __shared__ float s_h[4][2][50];
  int tid = threadIdx.y * 64 + threadIdx.x;
  for (int i = tid; i < 2500; i += 256) s_w2[i] = w2[i];
  if (tid < 50) { s_w1[tid] = w1[tid]; s_b1[tid] = b1[tid]; s_b2[tid] = b2[tid]; }
  __syncthreads();
  int row = blockIdx.x * 4 + threadIdx.y;
  if (threadIdx.x < 50 && row < nrow) {
    int j = threadIdx.x;
    for (int t = 0; t < 2; ++t) {
      float a = app[(size_t)row * 2 + t];
      float hv = a * s_w1[j] + s_b1[j];
      s_h[threadIdx.y][t][j] = hv > 0.0f ? hv : 0.0f;
    }
  }
  __syncthreads();
  if (row >= nrow) return;
  int c = threadIdx.x;
  float v = 0.0f;
  if (c < 50) {
    float acc = 2.0f * s_b2[c];
    for (int j = 0; j < 50; ++j)
      acc += (s_h[threadIdx.y][0][j] + s_h[threadIdx.y][1][j]) * s_w2[j * 50 + c];
    v = acc;
  }
  out[(size_t)row * 64 + c] = (_Float16)v;
}

// --------------------------- layernorm (wave/row) --------------------------
__global__ __launch_bounds__(256) void k_layernorm(
    const float* __restrict__ h, const float* __restrict__ g,
    const float* __restrict__ bb, _Float16* __restrict__ out, int nrow) {
  int lane = threadIdx.x & 31;
  int row  = (blockIdx.x * blockDim.x + threadIdx.x) >> 5;
  if (row >= nrow) return;
  const float* x = h + (size_t)row * 100;
  float v0 = x[lane], v1 = x[lane + 32], v2 = x[lane + 64];
  float v3 = (lane + 96 < 100) ? x[lane + 96] : 0.0f;
  float s = v0 + v1 + v2 + v3;
  for (int m = 1; m < 32; m <<= 1) s += __shfl_xor(s, m, 32);
  float mean = s * 0.01f;
  float d0 = v0 - mean, d1 = v1 - mean, d2 = v2 - mean;
  float d3 = (lane + 96 < 100) ? v3 - mean : 0.0f;
  float vs = d0 * d0 + d1 * d1 + d2 * d2 + d3 * d3;
  for (int m = 1; m < 32; m <<= 1) vs += __shfl_xor(vs, m, 32);
  float inv = rsqrtf(vs * 0.01f + 1e-5f);
  _Float16* o = out + (size_t)row * 128;
  o[lane]      = (_Float16)(d0 * inv * g[lane]      + bb[lane]);
  o[lane + 32] = (_Float16)(d1 * inv * g[lane + 32] + bb[lane + 32]);
  o[lane + 64] = (_Float16)(d2 * inv * g[lane + 64] + bb[lane + 64]);
  o[lane + 96] = (_Float16)((lane + 96 < 100)
                 ? (d3 * inv * g[lane + 96] + bb[lane + 96]) : 0.0f);
}

// ---------------------- f32 -> f16 cast with zero pad ----------------------
__global__ void k_cast_pad(const float* __restrict__ src, int lds_, int C,
                           _Float16* __restrict__ dst, int ldd, int nrow) {
  int i = blockIdx.x * blockDim.x + threadIdx.x;
  if (i >= nrow * ldd) return;
  int row = i / ldd, c = i % ldd;
  dst[i] = (_Float16)((c < C) ? src[(size_t)row * lds_ + c] : 0.0f);
}

__global__ void k_padzero(_Float16* __restrict__ buf, int ld, int c0, int nrow) {
  int w = ld - c0;
  int i = blockIdx.x * blockDim.x + threadIdx.x;
  if (i >= nrow * w) return;
  int row = i / w, c = c0 + i % w;
  buf[(size_t)row * ld + c] = (_Float16)0.0f;
}

// ----------------- depthwise causal conv (K=4) + SiLU ----------------------
__global__ void k_conv_silu(const float* __restrict__ xz,
                            const float* __restrict__ cw,
                            const float* __restrict__ cb,
                            float* __restrict__ xcf,
                            _Float16* __restrict__ xch, int nrow) {
  int i = blockIdx.x * blockDim.x + threadIdx.x;
  if (i >= nrow * 128) return;
  int row = i / 128, c = i % 128;
  if (c >= 100) { xch[(size_t)row * 128 + c] = (_Float16)0.0f; return; }
  int l = row & 511;
  float acc = cb[c];
#pragma unroll
  for (int k = 0; k < 4; ++k) {
    int dl = l + k - 3;
    if (dl >= 0) acc += xz[(size_t)(row + k - 3) * 200 + c] * cw[c * 4 + k];
  }
  float v = silu_f(acc);
  xcf[(size_t)row * 100 + c] = v;
  xch[(size_t)row * 128 + c] = (_Float16)v;
}

// ---------------- selective scan: lane-per-state, 2 chans/wave -------------
__global__ __launch_bounds__(256) void k_scan(
    const float* __restrict__ dtf, const float* __restrict__ xcf,
    const float* __restrict__ xdbl, const float* __restrict__ xz,
    const float* __restrict__ A_log, const float* __restrict__ Dv,
    _Float16* __restrict__ yh) {
  const int lane = threadIdx.x & 31;
  const int wave = threadIdx.x >> 5;
  const int s = lane & 15;
  const int d = blockIdx.y * 16 + wave * 2 + (lane >> 4);
  const int b = blockIdx.x;
  const bool active = (d < 100);
  float Aval = 0.0f, Dd = 0.0f;
  if (active) { Aval = -expf(A_log[d * 16 + s]); Dd = Dv[d]; }
  float hst = 0.0f;
  const size_t base = (size_t)b * 512;
  for (int l = 0; l < 512; ++l) {
    const size_t row = base + l;
    float dtv = 0.0f, xcv = 0.0f, Bs = 0.0f, Cs = 0.0f;
    if (active) {
      dtv = dtf[row * 100 + d];
      xcv = xcf[row * 100 + d];
      Bs  = xdbl[row * 48 + 7 + s];
      Cs  = xdbl[row * 48 + 23 + s];
    }
    hst = expf(dtv * Aval) * hst + dtv * Bs * xcv;
    float p = hst * Cs;
    p += __shfl_xor(p, 1, 32);
    p += __shfl_xor(p, 2, 32);
    p += __shfl_xor(p, 4, 32);
    p += __shfl_xor(p, 8, 32);
    if (active && s == 0) {
      float z = xz[row * 200 + 100 + d];
      float y = (p + xcv * Dd) * silu_f(z);
      yh[row * 128 + d] = (_Float16)y;
    }
  }
}

// ------------------- attention pooling (softmax over L) --------------------
__global__ __launch_bounds__(512) void k_pool(
    const float* __restrict__ h, const float* __restrict__ wagg,
    const float* __restrict__ waggb, _Float16* __restrict__ pooled) {
  __shared__ float att[512];
  __shared__ float red[512];
  const int b = blockIdx.x, t = threadIdx.x;
  const float* hr = h + ((size_t)b * 512 + t) * 100;
  float acc = waggb[0];
  for (int dd = 0; dd < 100; ++dd) acc += hr[dd] * wagg[dd];
  att[t] = acc; red[t] = acc;
  __syncthreads();
  for (int st = 256; st > 0; st >>= 1) {
    if (t < st) red[t] = fmaxf(red[t], red[t + st]);
    __syncthreads();
  }
  float mx = red[0];
  __syncthreads();
  float e = expf(att[t] - mx);
  att[t] = e; red[t] = e;
  __syncthreads();
  for (int st = 256; st > 0; st >>= 1) {
    if (t < st) red[t] += red[t + st];
    __syncthreads();
  }
  float inv = 1.0f / red[0];
  __syncthreads();
  att[t] *= inv;
  __syncthreads();
  if (t < 128) {
    float v = 0.0f;
    if (t < 100) {
      float s2 = 0.0f;
      for (int l = 0; l < 512; ++l)
        s2 += att[l] * h[((size_t)b * 512 + l) * 100 + t];
      v = s2;
    }
    pooled[b * 128 + t] = (_Float16)v;
  }
}

// ===========================================================================
extern "C" void kernel_launch(void* const* d_in, const int* in_sizes, int n_in,
                              void* d_out, int out_size, void* d_ws, size_t ws_size,
                              hipStream_t stream) {
  (void)in_sizes; (void)n_in; (void)out_size; (void)ws_size;
  constexpr int Bb = 100, Ll = 512, NR = Bb * Ll;     // 51200 rows (mult of 16)
  constexpr int MT = NR / 16;                         // 3200 row tiles
  auto F = [&](int i) { return (const float*)d_in[i]; };
  auto I = [&](int i) { return (const int*)d_in[i]; };

  // ---- input indices (setup_inputs dict order, params depth-first) ----
  // 0..8 runtime arrays; 9.. params
  const int P_NODEF = 9, P_EDGEF = 10, P_TW = 11, P_PNW = 12, P_PNB = 13,
            P_PEW = 14, P_PEB = 15, P_PTW = 16, P_PTB = 17, P_PFW = 18,
            P_PFB = 19, P_NW1 = 20, P_NB1 = 21, P_NW2 = 22, P_NB2 = 23,
            P_RW = 24, P_RB = 25, P_LG = 26, P_LB = 27, P_F1W = 28, P_F1B = 29,
            P_F2W = 30, P_F2B = 31, P_AGW = 32, P_AGB = 33, P_OW = 34, P_OB = 35,
            P_M0 = 36;                                 // + 9 per mamba layer

  // ------------------------- workspace layout ------------------------------
  char* base = (char*)d_ws;
  size_t off = 0;
  auto alloc = [&](size_t bytes) -> void* {
    void* p = base + off;
    off += (bytes + 255) & ~(size_t)255;
    return p;
  };
  auto ah = [&](size_t n) { return (_Float16*)alloc(n * 2); };
  auto af = [&](size_t n) { return (float*)alloc(n * 4); };

  _Float16 *WTnode = ah(64 * 192), *WTedge = ah(64 * 192), *WTtime = ah(64 * 128),
           *WTnif = ah(64 * 64), *WTred = ah(112 * 224), *WTf1 = ah(400 * 128),
           *WTf2 = ah(112 * 416), *WTout = ah(176 * 128);
  _Float16 *WTin[2], *WTxp[2], *WTdt[2], *WTop[2];
  for (int i = 0; i < 2; ++i) {
    WTin[i] = ah(208 * 128); WTxp[i] = ah(48 * 128);
    WTdt[i] = ah(112 * 32);  WTop[i] = ah(112 * 128);
  }
  _Float16 *ABUF = ah((size_t)NR * 192);   // gathered features (node/edge/time)
  _Float16 *CHH  = ah((size_t)NR * 224);   // concat proj (f16, K-padded)
  _Float16 *XH1  = ah((size_t)NR * 128);   // hn / h_h / xc_h / y_h
  _Float16 *XH2  = ah((size_t)NR * 416);   // ffn hidden / dt input
  _Float16 *NIFH = ah((size_t)NR * 64);
  _Float16 *POOLH = ah(112 * 128);
  float *APPS = af((size_t)NR * 2), *APPD = af((size_t)NR * 2);
  float *HB = af((size_t)NR * 100), *XZ = af((size_t)NR * 200),
        *XCF = af((size_t)NR * 100), *XDB = af((size_t)NR * 48),
        *DTF = af((size_t)NR * 100);

  const dim3 B256(256);
  auto g1 = [](int total) { return dim3((total + 255) / 256); };
  auto gg = [](int mt, int nt) { return dim3((mt * nt + 7) / 8); };

  // -------------------- once-per-call prep ---------------------------------
  (void)hipMemsetAsync(CHH, 0, (size_t)NR * 224 * 2, stream);   // K-pad zeros
  (void)hipMemsetAsync(POOLH, 0, (size_t)112 * 128 * 2, stream);// M-pad zeros

  auto prep = [&](const float* W, _Float16* WT, int K, int N, int Kp, int Np) {
    k_prep_wt<<<g1(Np * Kp), B256, 0, stream>>>(W, WT, K, N, Kp, Np);
  };
  prep(F(P_PNW), WTnode, 172, 50, 192, 64);
  prep(F(P_PEW), WTedge, 172, 50, 192, 64);
  prep(F(P_PTW), WTtime, 100, 50, 128, 64);
  prep(F(P_PFW), WTnif, 50, 50, 64, 64);
  prep(F(P_RW),  WTred, 200, 100, 224, 112);
  prep(F(P_F1W), WTf1, 100, 400, 128, 400);
  prep(F(P_F2W), WTf2, 400, 100, 416, 112);
  prep(F(P_OW),  WTout, 100, 172, 128, 176);
  for (int i = 0; i < 2; ++i) {
    int pb = P_M0 + i * 9;
    prep(F(pb + 0), WTin[i], 100, 200, 128, 208);   // in_proj
    prep(F(pb + 3), WTxp[i], 100, 39, 128, 48);     // x_proj
    prep(F(pb + 4), WTdt[i], 7, 100, 32, 112);      // dt_w
    prep(F(pb + 8), WTop[i], 100, 100, 128, 112);   // out_proj
  }

  // NIF counts (shared by both sides)
  k_nif_counts<<<dim3(Bb), dim3(512), 0, stream>>>(
      I(3), I(4), I(0), I(1), APPS, APPD);

  float* out = (float*)d_out;

  // ------------------------------ per side ---------------------------------
  for (int side = 0; side < 2; ++side) {
    const int*   nb  = I(3 + side);
    const int*   eb  = I(5 + side);
    const float* nbt = F(7 + side);
    const float* app = side ? APPD : APPS;

    // NIF encoder -> f16 (NR x 64)
    k_nif_enc<<<dim3(NR / 4), dim3(64, 4), 0, stream>>>(
        app, F(P_NW1), F(P_NB1), F(P_NW2), F(P_NB2), NIFH, NR);

    // node features -> proj -> ch[:,0:50]
    k_gather_cast<<<g1(NR * 192), B256, 0, stream>>>(F(P_NODEF), nb, ABUF, 172, 192, NR);
    gemm_wmma<0, 2><<<gg(MT, 4), B256, 0, stream>>>(
        ABUF, WTnode, 192, F(P_PNB), nullptr, CHH, 224, 0, NR, 50, MT, 4);
    // edge features -> proj -> ch[:,50:100]
    k_gather_cast<<<g1(NR * 192), B256, 0, stream>>>(F(P_EDGEF), eb, ABUF, 172, 192, NR);
    gemm_wmma<0, 2><<<gg(MT, 4), B256, 0, stream>>>(
        ABUF, WTedge, 192, F(P_PEB), nullptr, CHH, 224, 50, NR, 50, MT, 4);
    // time features -> proj -> ch[:,100:150]
    k_time<<<g1(NR * 128), B256, 0, stream>>>(F(2), nbt, F(P_TW), ABUF, NR);
    gemm_wmma<0, 2><<<gg(MT, 4), B256, 0, stream>>>(
        ABUF, WTtime, 128, F(P_PTB), nullptr, CHH, 224, 100, NR, 50, MT, 4);
    // nif features -> proj -> ch[:,150:200]
    gemm_wmma<0, 2><<<gg(MT, 4), B256, 0, stream>>>(
        NIFH, WTnif, 64, F(P_PFB), nullptr, CHH, 224, 150, NR, 50, MT, 4);

    // reduce: h = ch @ reduce_w + b   (f32)
    gemm_wmma<0, 0><<<gg(MT, 7), B256, 0, stream>>>(
        CHH, WTred, 224, F(P_RB), HB, nullptr, 100, 0, NR, 100, MT, 7);

    // layernorm -> hn (f16, pad 128)
    k_layernorm<<<dim3(NR / 8), B256, 0, stream>>>(HB, F(P_LG), F(P_LB), XH1, NR);
    // FFN: g = gelu(hn @ w1 + b1)
    gemm_wmma<1, 2><<<gg(MT, 25), B256, 0, stream>>>(
        XH1, WTf1, 128, F(P_F1B), nullptr, XH2, 416, 0, NR, 400, MT, 25);
    k_padzero<<<g1(NR * 16), B256, 0, stream>>>(XH2, 416, 400, NR);
    // h += g @ w2 + b2
    gemm_wmma<0, 1><<<gg(MT, 7), B256, 0, stream>>>(
        XH2, WTf2, 416, F(P_F2B), HB, nullptr, 100, 0, NR, 100, MT, 7);

    // ------------------------- mamba layers --------------------------------
    for (int li = 0; li < 2; ++li) {
      int pb = P_M0 + li * 9;
      // h -> f16
      k_cast_pad<<<g1(NR * 128), B256, 0, stream>>>(HB, 100, 100, XH1, 128, NR);
      // xz = h @ in_proj   (f32, NR x 200)
      gemm_wmma<0, 0><<<gg(MT, 13), B256, 0, stream>>>(
          XH1, WTin[li], 128, nullptr, XZ, nullptr, 200, 0, NR, 200, MT, 13);
      // causal depthwise conv + silu -> xcf (f32) + xc_h (f16 in XH1)
      k_conv_silu<<<g1(NR * 128), B256, 0, stream>>>(
          XZ, F(pb + 1), F(pb + 2), XCF, XH1, NR);
      // xdbl = xc @ x_proj  (f32, NR x 48; cols 0..38 valid)
      gemm_wmma<0, 0><<<gg(MT, 3), B256, 0, stream>>>(
          XH1, WTxp[li], 128, nullptr, XDB, nullptr, 48, 0, NR, 39, MT, 3);
      // dt = softplus(xdbl[:, :7] @ dt_w + dt_b)
      k_cast_pad<<<g1(NR * 32), B256, 0, stream>>>(XDB, 48, 7, XH2, 32, NR);
      gemm_wmma<2, 0><<<gg(MT, 7), B256, 0, stream>>>(
          XH2, WTdt[li], 32, F(pb + 5), DTF, nullptr, 100, 0, NR, 100, MT, 7);
      // selective scan + D skip + silu(z) gate -> y_h (f16 in XH1)
      k_scan<<<dim3(Bb, 7), B256, 0, stream>>>(
          DTF, XCF, XDB, XZ, F(pb + 6), F(pb + 7), XH1);
      // h += y @ out_proj
      gemm_wmma<0, 1><<<gg(MT, 7), B256, 0, stream>>>(
          XH1, WTop[li], 128, nullptr, HB, nullptr, 100, 0, NR, 100, MT, 7);
    }

    // attention pooling -> pooled_h (f16, 112 x 128)
    k_pool<<<dim3(Bb), dim3(512), 0, stream>>>(HB, F(P_AGW), F(P_AGB), POOLH);
    // emb = pooled @ out_w + out_b -> out[:, side*172 : side*172+172]
    gemm_wmma<0, 0><<<gg(7, 11), B256, 0, stream>>>(
        POOLH, WTout, 128, F(P_OB), out, nullptr, 344, side * 172, Bb, 172, 7, 11);
  }
}
